// LSTM_Operator_58377195487728
// MI455X (gfx1250) — matrix operand order repaired and checked by reading, not compiled, not run on hardware
//
#include <hip/hip_runtime.h>

// ---------------------------------------------------------------------------
// Types for CDNA5 WMMA
// ---------------------------------------------------------------------------
typedef __attribute__((ext_vector_type(16))) __bf16 v16bf;
typedef __attribute__((ext_vector_type(8)))  float  v8f;
typedef __attribute__((ext_vector_type(4)))  unsigned int ui4;
typedef __attribute__((ext_vector_type(2)))  float  f2;

struct BPair { ui4 lo, hi; };   // 32 bytes == v16bf

__device__ inline v16bf make_frag(const __bf16* p0, const __bf16* p1) {
    BPair t;
    t.lo = *(const ui4*)p0;   // 16B ds_load
    t.hi = *(const ui4*)p1;   // 16B ds_load
    return __builtin_bit_cast(v16bf, t);
}

__device__ inline v8f wmma_bf16(v16bf a, v16bf b, v8f c) {
    // D = A(16x32 bf16) * B(32x16 bf16) + C(16x16 f32)
    return __builtin_amdgcn_wmma_f32_16x16x32_bf16(
        /*neg_a=*/false, a, /*neg_b=*/false, b,
        /*c_mod=*/(short)0, c, /*reuse_a=*/false, /*reuse_b=*/false);
}

// float -> bf16 bits, round-to-nearest-even (branch-free)
__device__ inline unsigned bf16bits(float f) {
    unsigned u = __builtin_bit_cast(unsigned, f);
    return (u + 0x7FFFu + ((u >> 16) & 1u)) >> 16;
}
__device__ inline unsigned pack2(float a, float b) {
    return bf16bits(a) | (bf16bits(b) << 16);
}
__device__ inline __bf16 f2bf(float f) {
    unsigned short s = (unsigned short)bf16bits(f);
    return __builtin_bit_cast(__bf16, s);
}

__device__ inline float sigmoidf_(float x) { return 1.0f / (1.0f + __expf(-x)); }

// Problem constants
#define BATCH   16
#define TSTEPS  2048
#define DDIM    1024
#define NGATES  4096            // 4*DDIM
#define MROWS   32768           // BATCH*TSTEPS

// ---------------------------------------------------------------------------
// Kernel 0: init recurrent state (h0 bf16, c0 f32) and reset grid barrier
// ---------------------------------------------------------------------------
__global__ void lstm_init(const float* __restrict__ carry,
                          __bf16* __restrict__ hbuf0,
                          float*  __restrict__ cbuf,
                          unsigned* __restrict__ cnt) {
    int i = blockIdx.x * blockDim.x + threadIdx.x;   // 0..16383 == b*1024+d
    if (i < BATCH * DDIM) {
        hbuf0[i] = f2bf(carry[i * 2 + 0]);
        cbuf[i]  =      carry[i * 2 + 1];
    }
    if (i == 0) *cnt = 0u;
}

// ---------------------------------------------------------------------------
// Kernel 1: xg = x @ Wi   (M=32768, K=1024, N=4096), bf16 WMMA, f32 accum
// Block: 256 threads (8 waves as 4Mx2N), tile 128x64, BK=32.
// Double-buffered LDS; next K-tile prefetched into regs during WMMA.
// ---------------------------------------------------------------------------
__device__ inline void stage_load(const float* __restrict__ x,
                                  const float* __restrict__ Wi,
                                  int m0, int n0, int k0, int tid,
                                  unsigned apk[8], float wreg[8]) {
#pragma unroll
    for (int j = 0; j < 8; ++j) {           // x tile: 128x32 as 2048 f32 pairs
        int idx = tid + j * 256;
        int r = idx >> 4, c2 = idx & 15;
        f2 v = __builtin_nontemporal_load(
            (const f2*)&x[(size_t)(m0 + r) * DDIM + k0 + c2 * 2]);
        apk[j] = pack2(v.x, v.y);
    }
#pragma unroll
    for (int j = 0; j < 8; ++j) {           // Wi tile: 32x64
        int idx = tid + j * 256;
        int k = idx >> 6, n = idx & 63;
        wreg[j] = Wi[(size_t)(k0 + k) * NGATES + n0 + n];
    }
}

__device__ inline void stage_store(__bf16* __restrict__ sAb,
                                   __bf16* __restrict__ sBb, int tid,
                                   const unsigned apk[8], const float wreg[8]) {
#pragma unroll
    for (int j = 0; j < 8; ++j) {           // packed b32 LDS stores
        int idx = tid + j * 256;
        int r = idx >> 4, c2 = idx & 15;
        *(unsigned*)&sAb[r * 40 + c2 * 2] = apk[j];
    }
#pragma unroll
    for (int j = 0; j < 8; ++j) {           // transposed (col-major) scalar
        int idx = tid + j * 256;
        int k = idx >> 6, n = idx & 63;
        sBb[n * 40 + k] = f2bf(wreg[j]);
    }
}

__global__ __launch_bounds__(256) void lstm_gemm_xg(
    const float* __restrict__ x, const float* __restrict__ Wi,
    float* __restrict__ xg) {
    __shared__ __bf16 sA[2][128 * 40];   // pitch 40 halfs (K=32 + pad)
    __shared__ __bf16 sB[2][64 * 40];    // col-major, pitch 40 halfs

    const int tid   = threadIdx.x;
    const int lane  = tid & 31;
    const int wave  = tid >> 5;
    const int waveM = wave >> 1;      // 0..3
    const int waveN = wave & 1;       // 0..1
    const int m0    = blockIdx.y * 128;
    const int n0    = blockIdx.x * 64;
    const int hi    = lane >> 4;
    const int row   = lane & 15;

    unsigned apk[8];
    float    wreg[8];

    stage_load(x, Wi, m0, n0, 0, tid, apk, wreg);
    stage_store(&sA[0][0], &sB[0][0], tid, apk, wreg);
    __syncthreads();

    v8f C[2][2] = {};
#pragma unroll 1
    for (int kb = 0; kb < 32; ++kb) {
        const int cur = kb & 1;
        if (kb < 31)  // prefetch next K tile while WMMAs run
            stage_load(x, Wi, m0, n0, (kb + 1) * 32, tid, apk, wreg);
#pragma unroll
        for (int mi = 0; mi < 2; ++mi) {
            const __bf16* ap = &sA[cur][(waveM * 32 + mi * 16 + row) * 40 + hi * 8];
            v16bf A = make_frag(ap, ap + 16);
#pragma unroll
            for (int ni = 0; ni < 2; ++ni) {
                const __bf16* bp = &sB[cur][(waveN * 32 + ni * 16 + row) * 40 + hi * 8];
                v16bf B = make_frag(bp, bp + 16);
                C[mi][ni] = wmma_bf16(A, B, C[mi][ni]);
            }
        }
        if (kb < 31)
            stage_store(&sA[cur ^ 1][0], &sB[cur ^ 1][0], tid, apk, wreg);
        __syncthreads();
    }

#pragma unroll
    for (int mi = 0; mi < 2; ++mi)
#pragma unroll
        for (int ni = 0; ni < 2; ++ni)
#pragma unroll
            for (int r = 0; r < 8; ++r) {
                int grow = m0 + waveM * 32 + mi * 16 + r + hi * 8;
                int gcol = n0 + waveN * 32 + ni * 16 + row;
                __builtin_nontemporal_store(C[mi][ni][r],
                                            &xg[(size_t)grow * NGATES + gcol]);
            }
}

// ---------------------------------------------------------------------------
// Kernel 2: persistent recurrent scan. 32 workgroups x 256 threads.
// Workgroup wg owns d in [wg*32, wg*32+32): 8 gate tiles (4 gates x 2 halves),
// one 16x16 tile per wave. Entire Wh slice (256KB bf16) pinned in LDS.
// h broadcast staged via CDNA5 async global->LDS copies (ASYNCcnt).
// ---------------------------------------------------------------------------
__device__ inline void grid_barrier(unsigned* cnt, unsigned target) {
    __threadfence();
    __syncthreads();
    if (threadIdx.x == 0) {
        atomicAdd(cnt, 1u);
        while (atomicAdd(cnt, 0u) < target) { __builtin_amdgcn_s_sleep(1); }
    }
    __syncthreads();
    __threadfence();
}

__global__ __launch_bounds__(256) void lstm_recurrent(
    const float* __restrict__ Wh, const float* __restrict__ bias,
    const float* __restrict__ xg, float* __restrict__ out,
    float* __restrict__ carry_out,
    __bf16* __restrict__ hbuf0, __bf16* __restrict__ hbuf1,
    float* __restrict__ cbuf, unsigned* __restrict__ cnt) {

    // 256KB: Wh slice in B-fragment layout [tile8][kb32][lane32][16 halfs]
    __shared__ __bf16 sWh[8 * 32 * 32 * 16];
    // 33KB: current h, 16 rows x pitch 1032 halfs (2064B, 16B-aligned rows)
    __shared__ __bf16 sH[16 * 1032];
    // 8KB: gate tile exchange [tile8][vgpr8][lane32] f32
    __shared__ float sG[8 * 8 * 32];

    const int tid  = threadIdx.x;
    const int lane = tid & 31;
    const int wave = tid >> 5;                // tile id 0..7
    const int wg   = blockIdx.x;              // 0..31
    const int hi   = lane >> 4;
    const int row  = lane & 15;

    // ---- prologue: convert this WG's Wh slice into LDS fragments (one time)
    for (int idx = tid; idx < 8 * 32 * 32 * 16; idx += 256) {
        int i  = idx & 15;
        int ln = (idx >> 4) & 31;
        int kb = (idx >> 9) & 31;
        int w  = idx >> 14;
        int gate = w >> 1, half = w & 1;
        int col = gate * DDIM + wg * 32 + half * 16 + (ln & 15);
        int k   = kb * 32 + ((ln >> 4) << 3) + (i & 7) + ((i >> 3) << 4);  // lanes<16: K 0..7,16..23, else 8..15,24..31
        sWh[idx] = f2bf(Wh[(size_t)k * NGATES + col]);
    }
    // per-lane tile bias (constant for all steps)
    const int gate = wave >> 1, half = wave & 1;
    const int colBase = gate * DDIM + wg * 32 + half * 16;
    const float bv = bias[colBase + row];

    // per-thread async-copy addressing (8 x 16B chunks of the 32KB h image)
    const unsigned sHbase = (unsigned)(uintptr_t)(void*)&sH[0];
    __syncthreads();

    unsigned barTarget = 0;
#pragma unroll 1
    for (int t = 0; t < TSTEPS; ++t) {
        const __bf16* hcur  = (t & 1) ? hbuf1 : hbuf0;
        __bf16*       hnext = (t & 1) ? hbuf0 : hbuf1;

        // stage h (16x1024 bf16) into padded LDS via async global->LDS b128
        {
            unsigned long long gbase = (unsigned long long)(uintptr_t)hcur;
#pragma unroll
            for (int j = 0; j < 8; ++j) {
                int ch = tid + j * 256;          // 2048 x 16B chunks
                int r = ch >> 7, c16 = ch & 127;
                unsigned dst = sHbase + (unsigned)(r * 2064 + c16 * 16);
                unsigned long long src = gbase + (unsigned long long)(r * 2048 + c16 * 16);
                asm volatile("global_load_async_to_lds_b128 %0, %1, off"
                             :: "v"(dst), "v"(src) : "memory");
            }
            asm volatile("s_wait_asynccnt 0" ::: "memory");
        }
        __syncthreads();

        // ---- GEMM: one 16x16 gate tile per wave, K=1024 in 32 WMMA steps
        v8f C = {};
#pragma unroll
        for (int kb = 0; kb < 32; ++kb) {
            const __bf16* ap = &sH[row * 1032 + kb * 32 + hi * 8];
            v16bf A = make_frag(ap, ap + 16);
            const __bf16* bp = &sWh[((wave * 32 + kb) * 32 + lane) * 16];
            v16bf B = make_frag(bp, bp + 8);
            C = wmma_bf16(A, B, C);
        }

        // add xg + bias, publish pre-activation gates to LDS
        const int gcol = colBase + row;
#pragma unroll
        for (int r = 0; r < 8; ++r) {
            int bb = r + hi * 8;   // batch index of this C element
            float xv = __builtin_nontemporal_load(
                &xg[((size_t)(bb * TSTEPS + t)) * NGATES + gcol]);
            sG[(wave * 8 + r) * 32 + lane] = C[r] + bv + xv;
        }
        __syncthreads();

        // ---- elementwise LSTM cell: 16 batches x 32 local d = 512 elems
        for (int e = tid; e < 512; e += 256) {
            int bb = e >> 5;              // batch
            int dl = e & 31;              // local d
            int lidx  = (dl & 15) + ((bb & 8) ? 16 : 0);
            int r     = bb & 7;
            int thalf = dl >> 4;
            float gi = sG[((0 * 2 + thalf) * 8 + r) * 32 + lidx];
            float gf = sG[((1 * 2 + thalf) * 8 + r) * 32 + lidx];
            float gg = sG[((2 * 2 + thalf) * 8 + r) * 32 + lidx];
            float go = sG[((3 * 2 + thalf) * 8 + r) * 32 + lidx];
            int d = wg * 32 + dl;
            float c_old = cbuf[bb * DDIM + d];
            float c_new = sigmoidf_(gf) * c_old + sigmoidf_(gi) * tanhf(gg);
            float h_new = sigmoidf_(go) * tanhf(c_new);
            cbuf[bb * DDIM + d]  = c_new;
            hnext[bb * DDIM + d] = f2bf(h_new);
            __builtin_nontemporal_store(h_new,
                &out[((size_t)bb * TSTEPS + t) * DDIM + d]);
            if (t == TSTEPS - 1) {
                carry_out[(bb * DDIM + d) * 2 + 0] = h_new;
                carry_out[(bb * DDIM + d) * 2 + 1] = c_new;
            }
        }

        // one grid barrier per step (ping-pong h removes the second one)
        barTarget += 32;
        grid_barrier(cnt, barTarget);
    }
}

// ---------------------------------------------------------------------------
// Launch
// ---------------------------------------------------------------------------
extern "C" void kernel_launch(void* const* d_in, const int* in_sizes, int n_in,
                              void* d_out, int out_size, void* d_ws, size_t ws_size,
                              hipStream_t stream) {
    const float* x     = (const float*)d_in[0];   // (16,2048,1024)
    const float* carry = (const float*)d_in[1];   // (16,1024,2)
    const float* Wi    = (const float*)d_in[2];   // (1024,4096)
    const float* Wh    = (const float*)d_in[3];   // (1024,4096)
    const float* bias  = (const float*)d_in[4];   // (4096,)

    float* out       = (float*)d_out;                       // (16,2048,1024)
    float* carry_out = out + (size_t)BATCH * TSTEPS * DDIM; // (16,1024,2)

    char* ws = (char*)d_ws;
    const size_t XG_BYTES = (size_t)MROWS * NGATES * sizeof(float); // 512MB
    float*    xg    = (float*)ws;
    __bf16*   hbuf0 = (__bf16*)(ws + XG_BYTES);
    __bf16*   hbuf1 = hbuf0 + BATCH * DDIM;
    float*    cbuf  = (float*)(ws + XG_BYTES + 2 * BATCH * DDIM * sizeof(__bf16));
    unsigned* cnt   = (unsigned*)(ws + XG_BYTES +
                                  2 * BATCH * DDIM * sizeof(__bf16) +
                                  BATCH * DDIM * sizeof(float));

    lstm_init<<<(BATCH * DDIM + 255) / 256, 256, 0, stream>>>(carry, hbuf0, cbuf, cnt);
    lstm_gemm_xg<<<dim3(NGATES / 64, MROWS / 128), 256, 0, stream>>>(x, Wi, xg);
    lstm_recurrent<<<32, 256, 0, stream>>>(Wh, bias, xg, out, carry_out,
                                           hbuf0, hbuf1, cbuf, cnt);
}
